// Net_55868934586909
// MI455X (gfx1250) — compile-verified, matrix operand order checked by reading, not run on hardware
//
#include <hip/hip_runtime.h>
#include <math.h>

#define NNODES 50000
#define NEDGES 800000
#define DINDIM 256
#define HDIM   96
#define NLAYERS 4
#define NANCH  512
#define NCLS   40
#define NCLSP  48          // padded class dim (3x16)

typedef __attribute__((ext_vector_type(16))) __bf16 v16bf;
typedef __attribute__((ext_vector_type(8)))  float  v8f;

// ISA 7.12.2: for a 16-bit 16x32 A (or 32x16 B) fragment, the 16 elements held
// by a lane (half = lane>>4) are TWO CONTIGUOUS K-RUNS of 8:
//   e = 0..7  -> K = half*8 + e
//   e = 8..15 -> K = 16 + half*8 + (e-8)
// so fragments can be built from 128-bit loads, no per-element scatter.

__device__ __forceinline__ v8f wmma_bf16(v16bf a, v16bf b, v8f c) {
  return __builtin_amdgcn_wmma_f32_16x16x32_bf16(false, a, false, b, (short)0, c,
                                                 false, false);
}

__device__ __forceinline__ v16bf cvt16(float4 a, float4 b, float4 c, float4 d) {
  v16bf r;
  r[0]  = (__bf16)a.x; r[1]  = (__bf16)a.y; r[2]  = (__bf16)a.z; r[3]  = (__bf16)a.w;
  r[4]  = (__bf16)b.x; r[5]  = (__bf16)b.y; r[6]  = (__bf16)b.z; r[7]  = (__bf16)b.w;
  r[8]  = (__bf16)c.x; r[9]  = (__bf16)c.y; r[10] = (__bf16)c.z; r[11] = (__bf16)c.w;
  r[12] = (__bf16)d.x; r[13] = (__bf16)d.y; r[14] = (__bf16)d.z; r[15] = (__bf16)d.w;
  return r;
}

// A-fragment from a row-major fp32 row segment (p = row base + k0), 16B aligned.
__device__ __forceinline__ v16bf load_a(const float* __restrict__ p, int half) {
  const float* q = p + half * 8;
  return cvt16(*(const float4*)(q + 0),  *(const float4*)(q + 4),
               *(const float4*)(q + 16), *(const float4*)(q + 20));
}

__device__ __forceinline__ float4 mix4(float4 a, float4 b) {
  return make_float4(0.9f * a.x + 0.1f * b.x, 0.9f * a.y + 0.1f * b.y,
                     0.9f * a.z + 0.1f * b.z, 0.9f * a.w + 0.1f * b.w);
}
__device__ __forceinline__ float4 scl4(float4 a, float s) {
  return make_float4(a.x * s, a.y * s, a.z * s, a.w * s);
}

// ---- pack row-major [K][Nsrc] fp32 -> fragment-packed bf16 [KT][NT][lane][16]
__global__ void k_pack_b(const float* __restrict__ B, __bf16* __restrict__ out,
                         int K, int Nsrc, int Npad) {
  const int KT = K >> 5, NT = Npad >> 4;
  const int t = blockIdx.x * blockDim.x + threadIdx.x;
  if (t >= KT * NT * 32) return;
  const int lane = t & 31, frag = t >> 5;
  const int nt = frag % NT, kt = frag / NT;
  const int m = lane & 15, half = lane >> 4;
  const int n = nt * 16 + m, k0 = kt * 32 + half * 8;
  v16bf r;
#pragma unroll
  for (int j = 0; j < 8; ++j) {
    r[j]     = (n < Nsrc) ? (__bf16)B[(size_t)(k0 + j) * Nsrc + n]      : (__bf16)0.0f;
    r[8 + j] = (n < Nsrc) ? (__bf16)B[(size_t)(k0 + 16 + j) * Nsrc + n] : (__bf16)0.0f;
  }
  ((v16bf*)out)[t] = r;
}

// ---- pack normalized anchors (transposed gather) into fragment layout ------
// B[k][n] = hn[aidx[n]][k];  K=96 (3 kt), Npad=512 (32 nt)
__global__ void k_pack_anch(const float* __restrict__ hn, const int* __restrict__ aidx,
                            __bf16* __restrict__ out) {
  const int t = blockIdx.x * blockDim.x + threadIdx.x;
  if (t >= 3 * 32 * 32) return;
  const int lane = t & 31, frag = t >> 5;
  const int nt = frag % 32, kt = frag / 32;
  const int m = lane & 15, half = lane >> 4;
  const int row = aidx[nt * 16 + m];
  const float* p = hn + (size_t)row * HDIM + kt * 32 + half * 8;
  ((v16bf*)out)[t] = cvt16(*(const float4*)(p + 0),  *(const float4*)(p + 4),
                           *(const float4*)(p + 16), *(const float4*)(p + 20));
}

// ---------------- h0 = relu(x @ W_h + b_h) ; hcur = h0 --------------------
__global__ void __launch_bounds__(256)
k_hidden(const float* __restrict__ x, const __bf16* __restrict__ whp,
         const float* __restrict__ bh, float* __restrict__ h0,
         float* __restrict__ hcur) {
  __shared__ v16bf Bs[8 * 6 * 32];               // 48 KB fragment-packed
  const v16bf* src = (const v16bf*)whp;
  for (int i = threadIdx.x; i < 8 * 6 * 32; i += 256) Bs[i] = src[i];
  __syncthreads();

  const int lane = threadIdx.x & 31, wave = threadIdx.x >> 5;
  const int tile = blockIdx.x * 8 + wave;
  if (tile * 16 >= NNODES) return;
  const int row0 = tile * 16;
  const int m = lane & 15, half = lane >> 4;

  v8f acc[6] = {};
  const float* pa = x + (size_t)(row0 + m) * DINDIM;
#pragma unroll
  for (int kt = 0; kt < 8; ++kt) {
    const v16bf a = load_a(pa + kt * 32, half);
#pragma unroll
    for (int nt = 0; nt < 6; ++nt)
      acc[nt] = wmma_bf16(a, Bs[(kt * 6 + nt) * 32 + lane], acc[nt]);
  }
#pragma unroll
  for (int nt = 0; nt < 6; ++nt) {
    const int col = nt * 16 + m;
    const float bias = bh[col];
#pragma unroll
    for (int r = 0; r < 8; ++r) {
      const int row = row0 + r + half * 8;       // C/D: VGPR r -> M=r / r+8
      const float v = fmaxf(acc[nt][r] + bias, 0.0f);
      h0[(size_t)row * HDIM + col]   = v;
      hcur[(size_t)row * HDIM + col] = v;
    }
  }
}

// -------- agg[dst] += w * h[src]  (gather + native fp32 atomic scatter) ---
__global__ void k_aggregate(const float* __restrict__ h, const int* __restrict__ ei,
                            const float* __restrict__ ew, float* __restrict__ agg) {
  long long t = (long long)blockIdx.x * blockDim.x + threadIdx.x;
  if (t >= (long long)NEDGES * 24) return;
  const int e = (int)(t / 24), c = (int)(t % 24);
  const int s = ei[e], d = ei[NEDGES + e];
  const float w = ew[e];
  const float4 v = *((const float4*)(h + (size_t)s * HDIM) + c);
  float* out = agg + (size_t)d * HDIM + (size_t)c * 4;
  __hip_atomic_fetch_add(out + 0, v.x * w, __ATOMIC_RELAXED, __HIP_MEMORY_SCOPE_AGENT);
  __hip_atomic_fetch_add(out + 1, v.y * w, __ATOMIC_RELAXED, __HIP_MEMORY_SCOPE_AGENT);
  __hip_atomic_fetch_add(out + 2, v.z * w, __ATOMIC_RELAXED, __HIP_MEMORY_SCOPE_AGENT);
  __hip_atomic_fetch_add(out + 3, v.w * w, __ATOMIC_RELAXED, __HIP_MEMORY_SCOPE_AGENT);
}

// --- h = relu((1-beta)*hh + beta*(hh @ W_l)),  hh = 0.9*agg + 0.1*h0 ------
__global__ void __launch_bounds__(256)
k_layer(const float* __restrict__ agg, const float* __restrict__ h0,
        const __bf16* __restrict__ Wp, float* __restrict__ hout, float beta) {
  __shared__ v16bf Bs[3 * 6 * 32];               // 18 KB
  const v16bf* src = (const v16bf*)Wp;
  for (int i = threadIdx.x; i < 3 * 6 * 32; i += 256) Bs[i] = src[i];
  __syncthreads();

  const int lane = threadIdx.x & 31, wave = threadIdx.x >> 5;
  const int tile = blockIdx.x * 8 + wave;
  if (tile * 16 >= NNODES) return;
  const int row0 = tile * 16;
  const int m = lane & 15, half = lane >> 4;

  v8f acc[6] = {};
  const float* pa = agg + (size_t)(row0 + m) * HDIM;
  const float* p0 = h0  + (size_t)(row0 + m) * HDIM;
#pragma unroll
  for (int kt = 0; kt < 3; ++kt) {
    const float* qa = pa + kt * 32 + half * 8;
    const float* q0 = p0 + kt * 32 + half * 8;
    const v16bf a = cvt16(
        mix4(*(const float4*)(qa + 0),  *(const float4*)(q0 + 0)),
        mix4(*(const float4*)(qa + 4),  *(const float4*)(q0 + 4)),
        mix4(*(const float4*)(qa + 16), *(const float4*)(q0 + 16)),
        mix4(*(const float4*)(qa + 20), *(const float4*)(q0 + 20)));
#pragma unroll
    for (int nt = 0; nt < 6; ++nt)
      acc[nt] = wmma_bf16(a, Bs[(kt * 6 + nt) * 32 + lane], acc[nt]);
  }
#pragma unroll
  for (int nt = 0; nt < 6; ++nt) {
    const int col = nt * 16 + m;
#pragma unroll
    for (int r = 0; r < 8; ++r) {
      const size_t o = (size_t)(row0 + r + half * 8) * HDIM + col;
      const float hh = 0.9f * agg[o] + 0.1f * h0[o];
      hout[o] = fmaxf((1.0f - beta) * hh + beta * acc[nt][r], 0.0f);
    }
  }
}

// ---------------- hout = h @ fc_w + fc_b ----------------------------------
__global__ void __launch_bounds__(256)
k_fc(const float* __restrict__ hin, const __bf16* __restrict__ Wp,
     const float* __restrict__ bias, float* __restrict__ hout) {
  __shared__ v16bf Bs[3 * 6 * 32];
  const v16bf* src = (const v16bf*)Wp;
  for (int i = threadIdx.x; i < 3 * 6 * 32; i += 256) Bs[i] = src[i];
  __syncthreads();

  const int lane = threadIdx.x & 31, wave = threadIdx.x >> 5;
  const int tile = blockIdx.x * 8 + wave;
  if (tile * 16 >= NNODES) return;
  const int row0 = tile * 16;
  const int m = lane & 15, half = lane >> 4;

  v8f acc[6] = {};
  const float* pa = hin + (size_t)(row0 + m) * HDIM;
#pragma unroll
  for (int kt = 0; kt < 3; ++kt) {
    const v16bf a = load_a(pa + kt * 32, half);
#pragma unroll
    for (int nt = 0; nt < 6; ++nt)
      acc[nt] = wmma_bf16(a, Bs[(kt * 6 + nt) * 32 + lane], acc[nt]);
  }
#pragma unroll
  for (int nt = 0; nt < 6; ++nt) {
    const int col = nt * 16 + m;
    const float bb = bias[col];
#pragma unroll
    for (int r = 0; r < 8; ++r)
      hout[(size_t)(row0 + r + half * 8) * HDIM + col] = acc[nt][r] + bb;
  }
}

// -------------- L2-normalize rows of 96 (one wave32 per row) --------------
__global__ void k_rownorm96(const float* __restrict__ hin, float* __restrict__ hout) {
  const int row = blockIdx.x * 8 + (threadIdx.x >> 5);
  const int lane = threadIdx.x & 31;
  if (row >= NNODES) return;
  const float* p = hin + (size_t)row * HDIM;
  const float v0 = p[lane], v1 = p[lane + 32], v2 = p[lane + 64];
  float s = v0 * v0 + v1 * v1 + v2 * v2;
#pragma unroll
  for (int o = 16; o > 0; o >>= 1) s += __shfl_xor(s, o, 32);
  const float inv = 1.0f / fmaxf(sqrtf(s), 1e-12f);
  float* q = hout + (size_t)row * HDIM;
  q[lane] = v0 * inv; q[lane + 32] = v1 * inv; q[lane + 64] = v2 * inv;
}

// -------------- sims = hn @ anchors^T  (M=50000, K=96, N=512) -------------
__global__ void __launch_bounds__(256)
k_sims(const float* __restrict__ hn, const __bf16* __restrict__ anchp,
       float* __restrict__ sims) {
  __shared__ v16bf Bs[3 * 8 * 32];               // this block's 128-col slice, 24 KB
  const int cg = blockIdx.y;                     // 4 column groups of 128
  const v16bf* src = (const v16bf*)anchp;
  for (int i = threadIdx.x; i < 3 * 8 * 32; i += 256) {
    const int l = i & 31, f = i >> 5;
    const int ntl = f % 8, kt = f / 8;
    Bs[i] = src[(kt * 32 + cg * 8 + ntl) * 32 + l];
  }
  __syncthreads();

  const int lane = threadIdx.x & 31, wave = threadIdx.x >> 5;
  const int tile = blockIdx.x * 8 + wave;
  if (tile * 16 >= NNODES) return;
  const int row0 = tile * 16;
  const int m = lane & 15, half = lane >> 4;

  v8f acc[8] = {};
  const float* pa = hn + (size_t)(row0 + m) * HDIM;
#pragma unroll
  for (int kt = 0; kt < 3; ++kt) {
    const v16bf a = load_a(pa + kt * 32, half);
#pragma unroll
    for (int nt = 0; nt < 8; ++nt)
      acc[nt] = wmma_bf16(a, Bs[(kt * 8 + nt) * 32 + lane], acc[nt]);
  }
  const int c0 = cg * 128;
#pragma unroll
  for (int nt = 0; nt < 8; ++nt)
#pragma unroll
    for (int r = 0; r < 8; ++r)
      sims[(size_t)(row0 + r + half * 8) * NANCH + c0 + nt * 16 + m] = acc[nt][r];
}

// ------------- rinv[row] = 1/max(||sims[row]||, eps) ----------------------
__global__ void k_rnorm512(const float* __restrict__ sims, float* __restrict__ rinv) {
  const int row = blockIdx.x * 8 + (threadIdx.x >> 5);
  const int lane = threadIdx.x & 31;
  if (row >= NNODES) return;
  const float* p = sims + (size_t)row * NANCH;
  float s = 0.0f;
#pragma unroll
  for (int i = 0; i < 4; ++i) {
    const float4 v = *((const float4*)p + lane + i * 32);
    s += v.x * v.x + v.y * v.y + v.z * v.z + v.w * v.w;
  }
#pragma unroll
  for (int o = 16; o > 0; o >>= 1) s += __shfl_xor(s, o, 32);
  if (lane == 0) rinv[row] = 1.0f / fmaxf(sqrtf(s), 1e-12f);
}

// -------- logits = (sims * rinv) @ cls_w + cls_b  (cols padded 40 -> 48) --
__global__ void __launch_bounds__(256)
k_logits(const float* __restrict__ sims, const float* __restrict__ rinv,
         const __bf16* __restrict__ clsp, const float* __restrict__ clsb,
         float* __restrict__ logits) {
  __shared__ v16bf Bs[16 * 3 * 32];              // 48 KB
  const v16bf* src = (const v16bf*)clsp;
  for (int i = threadIdx.x; i < 16 * 3 * 32; i += 256) Bs[i] = src[i];
  __syncthreads();

  const int lane = threadIdx.x & 31, wave = threadIdx.x >> 5;
  const int tile = blockIdx.x * 8 + wave;
  if (tile * 16 >= NNODES) return;
  const int row0 = tile * 16;
  const int m = lane & 15, half = lane >> 4;

  v8f acc[3] = {};
  const float* pa = sims + (size_t)(row0 + m) * NANCH;
  const float ri = rinv[row0 + m];
#pragma unroll
  for (int kt = 0; kt < 16; ++kt) {
    const float* q = pa + kt * 32 + half * 8;
    const v16bf a = cvt16(scl4(*(const float4*)(q + 0),  ri),
                          scl4(*(const float4*)(q + 4),  ri),
                          scl4(*(const float4*)(q + 16), ri),
                          scl4(*(const float4*)(q + 20), ri));
#pragma unroll
    for (int nt = 0; nt < 3; ++nt)
      acc[nt] = wmma_bf16(a, Bs[(kt * 3 + nt) * 32 + lane], acc[nt]);
  }
#pragma unroll
  for (int nt = 0; nt < 3; ++nt) {
    const int col = nt * 16 + m;
    if (col < NCLS) {
      const float bb = clsb[col];
#pragma unroll
      for (int r = 0; r < 8; ++r)
        logits[(size_t)(row0 + r + half * 8) * NCLS + col] = acc[nt][r] + bb;
    }
  }
}

extern "C" void kernel_launch(void* const* d_in, const int* in_sizes, int n_in,
                              void* d_out, int out_size, void* d_ws, size_t ws_size,
                              hipStream_t stream) {
  const float* x     = (const float*)d_in[0];
  const float* ew    = (const float*)d_in[1];
  const float* wh    = (const float*)d_in[2];
  const float* bh    = (const float*)d_in[3];
  const float* convw = (const float*)d_in[4];
  const float* fcw   = (const float*)d_in[5];
  const float* fcb   = (const float*)d_in[6];
  const float* clsw  = (const float*)d_in[7];
  const float* clsb  = (const float*)d_in[8];
  const int*   ei    = (const int*)d_in[9];    // [2,E]: src then dst
  const int*   aidx  = (const int*)d_in[10];

  float* logits = (float*)d_out;
  float* sims   = (float*)d_out + (size_t)NNODES * NCLS;

  const size_t NH = (size_t)NNODES * HDIM;
  float*  h0   = (float*)d_ws;
  float*  hA   = h0 + NH;
  float*  hB   = hA + NH;
  float*  rinv = hB + NH;
  __bf16* whp   = (__bf16*)(rinv + NNODES);       // 8*6*512  = 24576
  __bf16* convp = whp   + 8 * 6 * 512;            // 4*3*6*512= 36864
  __bf16* fcp   = convp + 4 * 3 * 6 * 512;        // 3*6*512  = 9216
  __bf16* clsp  = fcp   + 3 * 6 * 512;            // 16*3*512 = 24576
  __bf16* anchp = clsp  + 16 * 3 * 512;           // 3*32*512 = 49152

  dim3 blk(256);
  const int tiles = (NNODES + 15) / 16;           // 3125
  const int gtile = (tiles + 7) / 8;              // 391 blocks x 8 waves

  // one-shot weight packing into WMMA fragment layout
  k_pack_b<<<6, blk, 0, stream>>>(wh, whp, DINDIM, HDIM, HDIM);
  for (int l = 0; l < NLAYERS; ++l)
    k_pack_b<<<3, blk, 0, stream>>>(convw + (size_t)l * HDIM * HDIM,
                                    convp + (size_t)l * 3 * 6 * 512,
                                    HDIM, HDIM, HDIM);
  k_pack_b<<<3, blk, 0, stream>>>(fcw, fcp, HDIM, HDIM, HDIM);
  k_pack_b<<<6, blk, 0, stream>>>(clsw, clsp, NANCH, NCLS, NCLSP);

  k_hidden<<<gtile, blk, 0, stream>>>(x, whp, bh, h0, hA);

  for (int l = 0; l < NLAYERS; ++l) {
    hipMemsetAsync(hB, 0, NH * sizeof(float), stream);
    const int gagg = (int)(((long long)NEDGES * 24 + 255) / 256);
    k_aggregate<<<gagg, blk, 0, stream>>>(hA, ei, ew, hB);
    const float beta = logf(0.5f / (float)(l + 1) + 1.0f);
    k_layer<<<gtile, blk, 0, stream>>>(hB, h0, convp + (size_t)l * 3 * 6 * 512,
                                       hA, beta);
  }

  k_fc<<<gtile, blk, 0, stream>>>(hA, fcp, fcb, hB);
  k_rownorm96<<<(NNODES + 7) / 8, blk, 0, stream>>>(hB, hA);   // hA = hn
  k_pack_anch<<<12, blk, 0, stream>>>(hA, aidx, anchp);

  dim3 gsim(gtile, 4);
  k_sims<<<gsim, blk, 0, stream>>>(hA, anchp, sims);
  k_rnorm512<<<(NNODES + 7) / 8, blk, 0, stream>>>(sims, rinv);
  k_logits<<<gtile, blk, 0, stream>>>(sims, rinv, clsp, clsb, logits);
}
